// MultiHeadedAttentionWithTypeScores_61546881352175
// MI455X (gfx1250) — compile-verified, matrix operand order checked by reading, not run on hardware
//
#include <hip/hip_runtime.h>

#define H  12
#define DM 768
#define S  2048
#define DK 64

typedef __attribute__((ext_vector_type(16))) __bf16 v16bf;
typedef __attribute__((ext_vector_type(8)))  __bf16 v8bf;
typedef __attribute__((ext_vector_type(8)))  float  v8f;
typedef int gv4i __attribute__((__vector_size__(16)));   // matches builtin param

#if __has_builtin(__builtin_amdgcn_global_load_async_to_lds_b128) && \
    __has_builtin(__builtin_amdgcn_s_wait_asynccnt)
#define USE_ASYNC_LDS 1
#else
#define USE_ASYNC_LDS 0
#endif

__device__ __forceinline__ unsigned short f2bf(float f) {
  union { float f; unsigned int u; } v; v.f = f;
  unsigned int u = v.u;
  u += 0x7FFFu + ((u >> 16) & 1u);   // round-to-nearest-even
  return (unsigned short)(u >> 16);
}

// Load a 16-element bf16 fragment for v_wmma_*_16x16x32_bf16 from an LDS row
// stored k-contiguous. `p` must point at row_base + 8*(lane>=16).
// ISA packing: elems 0..7 -> k = 8h+0..7 ; elems 8..15 -> k = 16+8h+0..7
__device__ __forceinline__ v16bf ld_frag(const unsigned short* p) {
  v8bf lo = *(const v8bf*)(p);
  v8bf hi = *(const v8bf*)(p + 16);
  v16bf r;
#pragma unroll
  for (int i = 0; i < 8; ++i) { r[i] = lo[i]; r[i + 8] = hi[i]; }
  return r;
}

__device__ __forceinline__ v8f wmma_bf16(v16bf a, v16bf b, v8f c) {
  return __builtin_amdgcn_wmma_f32_16x16x32_bf16(false, a, false, b,
                                                 (short)0, c, false, false);
}

// ---------------------------------------------------------------------------
// One-time fp32 -> bf16 conversion (8 elements / thread, b128 in / b128 out)
// ---------------------------------------------------------------------------
__global__ __launch_bounds__(256) void cvt_f32_bf16_kernel(
    const float* __restrict__ in, unsigned short* __restrict__ out, int n8)
{
  int i = blockIdx.x * 256 + threadIdx.x;
  if (i >= n8) return;
  float4 a = ((const float4*)in)[i * 2];
  float4 b = ((const float4*)in)[i * 2 + 1];
  union { unsigned short s[8]; uint4 v; } o;
  o.s[0] = f2bf(a.x); o.s[1] = f2bf(a.y); o.s[2] = f2bf(a.z); o.s[3] = f2bf(a.w);
  o.s[4] = f2bf(b.x); o.s[5] = f2bf(b.y); o.s[6] = f2bf(b.z); o.s[7] = f2bf(b.w);
  ((uint4*)out)[i] = o.v;
}

// ---------------------------------------------------------------------------
// C = (A[M,K] * W[N,K]^T + bias) * out_scale     (bf16 A/W, f32 accumulate)
// 256 threads = 8 waves; block tile 64(M) x 256(N); wave tile 32x64;
// 8 v_wmma per K-step per wave. OUT_TRANS stores C^T ([N][M]).
// ---------------------------------------------------------------------------
template <bool OUT_BF16, bool OUT_TRANS>
__global__ __launch_bounds__(256) void gemm_bf16_nt_kernel(
    const unsigned short* __restrict__ A, const unsigned short* __restrict__ W,
    const float* __restrict__ bias, void* __restrict__ Cout,
    int M, int N, int K, float out_scale)
{
  __shared__ unsigned short As[64 * 32];    // [m][k]
  __shared__ unsigned short Bs[256 * 32];   // [n][k]

  const int tid  = threadIdx.x;
  const int wv   = tid >> 5;
  const int lane = tid & 31;
  const int hh   = lane >> 4;
  const int lr   = lane & 15;
  const int bm   = blockIdx.y * 64;
  const int bn   = blockIdx.x * 256;
  const int wm   = (wv >> 2) * 32;
  const int wn   = (wv & 3) * 64;

  v8f acc[8] = {};

  const int ar = (tid * 8) >> 5, ak = (tid * 8) & 31;
  for (int k0 = 0; k0 < K; k0 += 32) {
    *(uint4*)&As[ar * 32 + ak] =
        *(const uint4*)&A[(size_t)(bm + ar) * K + k0 + ak];
#pragma unroll
    for (int j = 0; j < 4; ++j) {
      int i = (tid + j * 256) * 8;
      int br = i >> 5, bk = i & 31;
      *(uint4*)&Bs[br * 32 + bk] =
          *(const uint4*)&W[(size_t)(bn + br) * K + k0 + bk];
    }
    __syncthreads();
    v16bf a0 = ld_frag(As + (wm + lr) * 32 + hh * 8);
    v16bf a1 = ld_frag(As + (wm + 16 + lr) * 32 + hh * 8);
#pragma unroll
    for (int t = 0; t < 4; ++t) {
      v16bf b = ld_frag(Bs + (wn + t * 16 + lr) * 32 + hh * 8);
      acc[t]     = wmma_bf16(a0, b, acc[t]);
      acc[4 + t] = wmma_bf16(a1, b, acc[4 + t]);
    }
    __syncthreads();
  }

#pragma unroll
  for (int t = 0; t < 4; ++t) {
    const int n  = bn + wn + t * 16 + lr;
    const float bv = bias[n];
#pragma unroll
    for (int mt = 0; mt < 2; ++mt) {
#pragma unroll
      for (int r = 0; r < 8; ++r) {
        const int m = bm + wm + mt * 16 + r + 8 * hh;
        const float val = (acc[mt * 4 + t][r] + bv) * out_scale;
        const size_t idx = OUT_TRANS ? ((size_t)n * M + m) : ((size_t)m * N + n);
        if (OUT_BF16) ((unsigned short*)Cout)[idx] = f2bf(val);
        else          ((float*)Cout)[idx] = val;
      }
    }
  }
}

// ---------------------------------------------------------------------------
// Flash attention, post-softmax type_scores weighting.
// q,k bf16 [S][DM]; vT bf16 [DM][S]; ctx out bf16 [S][DM].
// Grid (S/64, H); 128 threads = 4 waves; wave owns 16 query rows.
// ---------------------------------------------------------------------------
__global__ __launch_bounds__(128) void attn_flash_kernel(
    const unsigned short* __restrict__ q, const unsigned short* __restrict__ k,
    const unsigned short* __restrict__ vT, const float* __restrict__ ts,
    const int* __restrict__ mask, unsigned short* __restrict__ ctx)
{
  __shared__ unsigned short Qs[64 * DK];     // [qlocal][d] (scale pre-folded)
  __shared__ unsigned short Ks[32 * DK];     // [kk][d]
  __shared__ unsigned short Vt[DK * 32];     // [d][kk]
  __shared__ unsigned short Ps[4 * 16 * 32]; // per-wave [q16][kk32]

  const int tid  = threadIdx.x;
  const int wv   = tid >> 5;
  const int lane = tid & 31;
  const int hh   = lane >> 4;
  const int lr   = lane & 15;
  const int h    = blockIdx.y;
  const int q0   = blockIdx.x * 64;
  const int col0 = h * DK;

#pragma unroll
  for (int j = 0; j < 4; ++j) {
    int i = (tid + j * 128) * 8;
    int row = i >> 6, d = i & 63;
    *(uint4*)&Qs[row * 64 + d] =
        *(const uint4*)&q[(size_t)(q0 + row) * DM + col0 + d];
  }
  __syncthreads();

  const v16bf qf0 = ld_frag(Qs + (wv * 16 + lr) * DK + hh * 8);
  const v16bf qf1 = ld_frag(Qs + (wv * 16 + lr) * DK + 32 + hh * 8);

  v8f c0 = {}, c1 = {}, c2 = {}, c3 = {};
  float mrow[8], lrow[8];
#pragma unroll
  for (int r = 0; r < 8; ++r) { mrow[r] = -1e30f; lrow[r] = 0.0f; }

  for (int kb = 0; kb < S; kb += 32) {
    __syncthreads();
    // ---- stage K tile [32][64] and V^T tile [64][32] (raw bf16 copies) ----
#if USE_ASYNC_LDS
#pragma unroll
    for (int j = 0; j < 2; ++j) {
      int c = tid + j * 128;                       // 16B chunk id, 0..255
      int kk = c >> 3, d = (c & 7) * 8;
      __builtin_amdgcn_global_load_async_to_lds_b128(
          (gv4i*)&k[(size_t)(kb + kk) * DM + col0 + d],
          (gv4i*)&Ks[kk * DK + d], 0, 0);
      int d2 = c >> 2, kk2 = (c & 3) * 8;
      __builtin_amdgcn_global_load_async_to_lds_b128(
          (gv4i*)&vT[(size_t)(col0 + d2) * S + kb + kk2],
          (gv4i*)&Vt[d2 * 32 + kk2], 0, 0);
    }
    __builtin_amdgcn_s_wait_asynccnt(0);
#else
#pragma unroll
    for (int j = 0; j < 2; ++j) {
      int c = tid + j * 128;
      int kk = c >> 3, d = (c & 7) * 8;
      *(uint4*)&Ks[kk * DK + d] =
          *(const uint4*)&k[(size_t)(kb + kk) * DM + col0 + d];
      int d2 = c >> 2, kk2 = (c & 3) * 8;
      *(uint4*)&Vt[d2 * 32 + kk2] =
          *(const uint4*)&vT[(size_t)(col0 + d2) * S + kb + kk2];
    }
#endif
    __syncthreads();

    // ---- scores: 16 queries x 32 keys, d=64 in two WMMA K-steps ----
    v8f s0 = {}, s1 = {};
    {
      v16bf b0 = ld_frag(Ks + lr * DK + hh * 8);
      v16bf b1 = ld_frag(Ks + (16 + lr) * DK + hh * 8);
      s0 = wmma_bf16(qf0, b0, s0);
      s1 = wmma_bf16(qf0, b1, s1);
      b0 = ld_frag(Ks + lr * DK + 32 + hh * 8);
      b1 = ld_frag(Ks + (16 + lr) * DK + 32 + hh * 8);
      s0 = wmma_bf16(qf1, b0, s0);
      s1 = wmma_bf16(qf1, b1, s1);
    }

    if (kb + 32 < S)
      __builtin_prefetch(ts + ((size_t)h * S + (size_t)(q0 + wv * 16)) * S + kb + 32, 0, 0);

    // ---- online softmax; numerator weight = exp * type_score ----
#pragma unroll
    for (int r = 0; r < 8; ++r) {
      const int qrow = q0 + wv * 16 + r + 8 * hh;
      const int key0 = kb + lr, key1 = kb + 16 + lr;
      float sa = s0[r], sb = s1[r];
      if (mask[(size_t)qrow * S + key0] == 0) sa = -1e9f;
      if (mask[(size_t)qrow * S + key1] == 0) sb = -1e9f;

      float vmax = fmaxf(sa, sb);
#pragma unroll
      for (int off = 1; off < 16; off <<= 1)
        vmax = fmaxf(vmax, __shfl_xor(vmax, off, 32));
      const float mnew = fmaxf(mrow[r], vmax);
      const float corr = __expf(mrow[r] - mnew);
      mrow[r] = mnew;
      lrow[r] *= corr;
      c0[r] *= corr; c1[r] *= corr; c2[r] *= corr; c3[r] *= corr;

      const float pa = __expf(sa - mnew);
      const float pb = __expf(sb - mnew);
      float rsum = pa + pb;
#pragma unroll
      for (int off = 1; off < 16; off <<= 1)
        rsum += __shfl_xor(rsum, off, 32);
      lrow[r] += rsum;

      const float ta = ts[((size_t)h * S + qrow) * S + key0];
      const float tb = ts[((size_t)h * S + qrow) * S + key1];
      const int ql = r + 8 * hh;
      Ps[(wv * 16 + ql) * 32 + lr]      = f2bf(pa * ta);
      Ps[(wv * 16 + ql) * 32 + 16 + lr] = f2bf(pb * tb);
    }

    // ---- numerator += P(16x32) * V(32x64) ----
    {
      v16bf ap = ld_frag(Ps + (wv * 16 + lr) * 32 + hh * 8);
      v16bf b;
      b  = ld_frag(Vt + (0 * 16 + lr) * 32 + hh * 8);
      c0 = wmma_bf16(ap, b, c0);
      b  = ld_frag(Vt + (1 * 16 + lr) * 32 + hh * 8);
      c1 = wmma_bf16(ap, b, c1);
      b  = ld_frag(Vt + (2 * 16 + lr) * 32 + hh * 8);
      c2 = wmma_bf16(ap, b, c2);
      b  = ld_frag(Vt + (3 * 16 + lr) * 32 + hh * 8);
      c3 = wmma_bf16(ap, b, c3);
    }
  }

#pragma unroll
  for (int r = 0; r < 8; ++r) {
    const int qrow = q0 + wv * 16 + r + 8 * hh;
    const float inv = 1.0f / (lrow[r] + 1e-30f);
    ctx[(size_t)qrow * DM + col0 + 0 * 16 + lr] = f2bf(c0[r] * inv);
    ctx[(size_t)qrow * DM + col0 + 1 * 16 + lr] = f2bf(c1[r] * inv);
    ctx[(size_t)qrow * DM + col0 + 2 * 16 + lr] = f2bf(c2[r] * inv);
    ctx[(size_t)qrow * DM + col0 + 3 * 16 + lr] = f2bf(c3[r] * inv);
  }
}

// ---------------------------------------------------------------------------
extern "C" void kernel_launch(void* const* d_in, const int* in_sizes, int n_in,
                              void* d_out, int out_size, void* d_ws, size_t ws_size,
                              hipStream_t stream) {
  (void)in_sizes; (void)n_in; (void)out_size; (void)ws_size;
  const float* query = (const float*)d_in[0];
  const float* key   = (const float*)d_in[1];
  const float* value = (const float*)d_in[2];
  const float* tsc   = (const float*)d_in[3];
  const int*   mask  = (const int*)  d_in[4];
  const float* Wq = (const float*)d_in[5];  const float* bq = (const float*)d_in[6];
  const float* Wk = (const float*)d_in[7];  const float* bk = (const float*)d_in[8];
  const float* Wv = (const float*)d_in[9];  const float* bv = (const float*)d_in[10];
  const float* Wo = (const float*)d_in[11]; const float* bo = (const float*)d_in[12];
  float* out = (float*)d_out;

  // workspace layout (all bf16 = unsigned short)
  unsigned short* p   = (unsigned short*)d_ws;
  unsigned short* xq  = p; p += (size_t)S * DM;   // inputs converted to bf16
  unsigned short* xk  = p; p += (size_t)S * DM;
  unsigned short* xv  = p; p += (size_t)S * DM;
  unsigned short* wqb = p; p += (size_t)DM * DM;  // weights converted to bf16
  unsigned short* wkb = p; p += (size_t)DM * DM;
  unsigned short* wvb = p; p += (size_t)DM * DM;
  unsigned short* wob = p; p += (size_t)DM * DM;
  unsigned short* qb  = p; p += (size_t)S * DM;   // projected q (pre-scaled)
  unsigned short* kb  = p; p += (size_t)S * DM;   // projected k
  unsigned short* vtb = p; p += (size_t)DM * S;   // projected v, transposed
  unsigned short* ctb = p;                        // attention context

  const int n8_in = (S * DM) / 8, n8_w = (DM * DM) / 8;
  cvt_f32_bf16_kernel<<<n8_in / 256, 256, 0, stream>>>(query, xq, n8_in);
  cvt_f32_bf16_kernel<<<n8_in / 256, 256, 0, stream>>>(key,   xk, n8_in);
  cvt_f32_bf16_kernel<<<n8_in / 256, 256, 0, stream>>>(value, xv, n8_in);
  cvt_f32_bf16_kernel<<<n8_w  / 256, 256, 0, stream>>>(Wq, wqb, n8_w);
  cvt_f32_bf16_kernel<<<n8_w  / 256, 256, 0, stream>>>(Wk, wkb, n8_w);
  cvt_f32_bf16_kernel<<<n8_w  / 256, 256, 0, stream>>>(Wv, wvb, n8_w);
  cvt_f32_bf16_kernel<<<n8_w  / 256, 256, 0, stream>>>(Wo, wob, n8_w);

  dim3 gg(DM / 256, S / 64);   // (3, 32)
  gemm_bf16_nt_kernel<true,  false><<<gg, 256, 0, stream>>>(xq, wqb, bq, qb,  S, DM, DM, 0.125f);
  gemm_bf16_nt_kernel<true,  false><<<gg, 256, 0, stream>>>(xk, wkb, bk, kb,  S, DM, DM, 1.0f);
  gemm_bf16_nt_kernel<true,  true ><<<gg, 256, 0, stream>>>(xv, wvb, bv, vtb, S, DM, DM, 1.0f);

  attn_flash_kernel<<<dim3(S / 64, H), 128, 0, stream>>>(qb, kb, vtb, tsc, mask, ctb);

  gemm_bf16_nt_kernel<false, false><<<gg, 256, 0, stream>>>(ctb, wob, bo, out, S, DM, DM, 1.0f);
}
